// MultiHeadAttention_85177791414455
// MI455X (gfx1250) — matrix operand order repaired and checked by reading, not compiled, not run on hardware
//
#include <hip/hip_runtime.h>

#define N_EMBED 1024
#define N_HEADS 16
#define HEAD_DIM 64
#define BATCH 4
#define SEQ 2048
#define MROWS (BATCH * SEQ) // 8192

typedef __bf16 bf16_t;
typedef bf16_t v16bf __attribute__((ext_vector_type(16)));
typedef bf16_t v2bf  __attribute__((ext_vector_type(2)));
typedef float  v8f   __attribute__((ext_vector_type(8)));
typedef unsigned int v4u __attribute__((ext_vector_type(4)));

// f32 -> bf16 (RNE). Prefer the hardware packed convert when the builtin exists.
#if __has_builtin(__builtin_amdgcn_cvt_pk_bf16_f32)
__device__ __forceinline__ unsigned int f2bf_pk(float lo, float hi) {
  union { v2bf b; unsigned int u; } c;
  c.b = __builtin_amdgcn_cvt_pk_bf16_f32(lo, hi);
  return c.u;
}
__device__ __forceinline__ unsigned short f2bf(float f) {
  return (unsigned short)(f2bf_pk(f, 0.f) & 0xFFFFu);
}
#else
__device__ __forceinline__ unsigned short f2bf(float f) {
  unsigned int u = __float_as_uint(f);
  u += 0x7FFFu + ((u >> 16) & 1u);
  return (unsigned short)(u >> 16);
}
__device__ __forceinline__ unsigned int f2bf_pk(float lo, float hi) {
  return (unsigned int)f2bf(lo) | ((unsigned int)f2bf(hi) << 16);
}
#endif

__device__ __forceinline__ float fast_exp2(float x) {
  return __builtin_amdgcn_exp2f(x);   // bare v_exp_f32
}

__device__ __forceinline__ v8f zero8() {
  v8f z = {0.f, 0.f, 0.f, 0.f, 0.f, 0.f, 0.f, 0.f};
  return z;
}

// A fragment 16x32 bf16 from [M][K] tile (stride in halfs, multiple of 8).
__device__ __forceinline__ v16bf load_frag_a(const unsigned short* base, int stride, int lane) {
  int m  = lane & 15;
  int kb = (lane >> 4) << 3;
  const unsigned short* p = base + m * stride + kb;
  union { v16bf bf; v4u u[2]; } r;
  r.u[0] = *(const v4u*)(p);
  r.u[1] = *(const v4u*)(p + 16);
  return r.bf;
}

// B fragment 32x16 bf16 from [N][K] tile. Lane half h holds K 8h..8h+7 and 16+8h..16+8h+7.
__device__ __forceinline__ v16bf load_frag_b(const unsigned short* base, int stride, int lane) {
  int n  = lane & 15;
  int kb = (lane >> 4) << 3;
  const unsigned short* p = base + n * stride + kb;
  union { v16bf bf; v4u u[2]; } r;
  r.u[0] = *(const v4u*)(p);
  r.u[1] = *(const v4u*)(p + 16);
  return r.bf;
}

__device__ __forceinline__ v8f wmma_bf16(v16bf a, v16bf b, v8f c) {
  return __builtin_amdgcn_wmma_f32_16x16x32_bf16(false, a, false, b, (short)0, c, false, false);
}

// ---------------------------------------------------------------------------
// Kernel 1: qkv = x @ W_attn + b_attn -> bf16 Q (pre-scaled log2e/8), K, V [B,H,L,64]
// Double-buffered LDS; one barrier per K-step.
// ---------------------------------------------------------------------------
__global__ __launch_bounds__(256) void qkv_gemm_kernel(
    const float* __restrict__ X, const float* __restrict__ W, const float* __restrict__ bias,
    unsigned short* __restrict__ Qo, unsigned short* __restrict__ Ko, unsigned short* __restrict__ Vo) {
  const int ldw = 3 * N_EMBED;
  int n0 = blockIdx.x * 128;
  int m0 = blockIdx.y * 128;
  int t = threadIdx.x, lane = t & 31, wave = t >> 5;
  int wm = wave & 3, wn = wave >> 2;
  __shared__ __align__(16) unsigned short sA[2][128 * 40];
  __shared__ __align__(16) unsigned short sB[2][128 * 40];
  v8f acc[2][4];
  for (int i = 0; i < 2; ++i)
    for (int j = 0; j < 4; ++j) acc[i][j] = zero8();

  auto load_tile = [&](int k0, int buf) {
    {  // A tile: 128x32 f32 -> bf16 packed b32 stores
      int row = t >> 1, ko = (t & 1) * 16;
      const float* src = X + (size_t)(m0 + row) * N_EMBED + k0 + ko;
      unsigned int* dst = (unsigned int*)(sA[buf] + row * 40 + ko);
#pragma unroll
      for (int i = 0; i < 8; ++i) dst[i] = f2bf_pk(src[2 * i], src[2 * i + 1]);
    }
    {  // B tile: W[k][n] -> LDS [n][k]; each thread owns one n, 16 contiguous k
      int nn = t & 127, kb = (t >> 7) * 16;
      const float* wsrc = W + (size_t)(k0 + kb) * ldw + n0 + nn;
      unsigned int* dst = (unsigned int*)(sB[buf] + nn * 40 + kb);
#pragma unroll
      for (int i = 0; i < 8; ++i)
        dst[i] = f2bf_pk(wsrc[(size_t)(2 * i) * ldw], wsrc[(size_t)(2 * i + 1) * ldw]);
    }
  };

  load_tile(0, 0);
  int cur = 0;
  for (int k0 = 0; k0 < N_EMBED; k0 += 32) {
    __syncthreads();                          // buf[cur] visible; buf[cur^1] free
    if (k0 + 32 < N_EMBED) load_tile(k0 + 32, cur ^ 1);
    v16bf a0 = load_frag_a(sA[cur] + (wm * 32) * 40, 40, lane);
    v16bf a1 = load_frag_a(sA[cur] + (wm * 32 + 16) * 40, 40, lane);
#pragma unroll
    for (int j = 0; j < 4; ++j) {
      v16bf bfr = load_frag_b(sB[cur] + (wn * 64 + j * 16) * 40, 40, lane);
      acc[0][j] = wmma_bf16(a0, bfr, acc[0][j]);
      acc[1][j] = wmma_bf16(a1, bfr, acc[1][j]);
    }
    cur ^= 1;
  }

  // epilogue: sel is BLOCK-UNIFORM (128 | 1024) -> no per-element branching
  int sel = n0 >> 10;                                  // 0=q 1=k 2=v
  unsigned short* dst = (sel == 0) ? Qo : (sel == 1) ? Ko : Vo;
  // Fold 1/sqrt(64) AND log2(e) into Q so attention softmax can use exp2.
  float scale = (sel == 0) ? (0.125f * 1.4426950408889634f) : 1.0f;
  int b = m0 >> 11;                                    // batch uniform per block
  float bias_j[4];
  size_t colbase[4];
#pragma unroll
  for (int j = 0; j < 4; ++j) {
    int n = n0 + wn * 64 + j * 16 + (lane & 15);
    bias_j[j] = bias[n];
    int d = n & 1023, h = d >> 6, hd = d & 63;
    colbase[j] = ((size_t)(b * N_HEADS + h)) * SEQ * HEAD_DIM + hd;
  }
#pragma unroll
  for (int mi = 0; mi < 2; ++mi)
#pragma unroll
    for (int j = 0; j < 4; ++j)
#pragma unroll
      for (int r = 0; r < 8; ++r) {
        int m = m0 + wm * 32 + mi * 16 + r + ((lane >> 4) << 3);
        int l = m & 2047;
        dst[colbase[j] + (size_t)l * HEAD_DIM] = f2bf((acc[mi][j][r] + bias_j[j]) * scale);
      }
}

// ---------------------------------------------------------------------------
// Kernel 2: causal flash attention per (q-tile of 128, b*h). 8 waves x 16 rows.
// Double-buffered K/V tiles; wave-private P round trip uses wavefront fence.
// Scores are in the log2 domain (log2e folded into Q) -> bare v_exp_f32.
// ---------------------------------------------------------------------------
__global__ __launch_bounds__(256) void attn_kernel(
    const unsigned short* __restrict__ Q, const unsigned short* __restrict__ K,
    const unsigned short* __restrict__ V, unsigned short* __restrict__ O) {
  int qt = blockIdx.x, bh = blockIdx.y;
  int b = bh >> 4, h = bh & 15;
  int q0 = qt * 128;
  int t = threadIdx.x, lane = t & 31, wave = t >> 5;
  const unsigned short* Qg = Q + (size_t)bh * SEQ * HEAD_DIM;
  const unsigned short* Kg = K + (size_t)bh * SEQ * HEAD_DIM;
  const unsigned short* Vg = V + (size_t)bh * SEQ * HEAD_DIM;

  __shared__ __align__(16) unsigned short sK[2][64 * 72];    // [key][hd]
  __shared__ __align__(16) unsigned short sV[2][64 * 72];    // [hd][key]
  __shared__ __align__(16) unsigned short sP[8][16 * 72];    // per-wave P tiles

  const unsigned short* qbase = Qg + (size_t)(q0 + wave * 16) * HEAD_DIM;
  v16bf qfr0 = load_frag_a(qbase + 0, HEAD_DIM, lane);
  v16bf qfr1 = load_frag_a(qbase + 32, HEAD_DIM, lane);

  v8f o_acc[4];
  for (int j = 0; j < 4; ++j) o_acc[j] = zero8();
  float m_r[8], l_r[8];
#pragma unroll
  for (int r = 0; r < 8; ++r) { m_r[r] = -3.0e38f; l_r[r] = 0.f; }

  int qrow_base = q0 + wave * 16 + ((lane >> 4) << 3);
  int nkv = (q0 >> 6) + 2;   // covers keys up to q0+127

  auto load_kv = [&](int kt, int buf) {
    int k0 = kt * 64;
    int row = t >> 2, off = (t & 3) * 16;
    {  // K: straight b128 copies
      const unsigned short* src = Kg + (size_t)(k0 + row) * HEAD_DIM + off;
      unsigned short* dk = sK[buf] + row * 72 + off;
      *(v4u*)(dk)     = *(const v4u*)(src);
      *(v4u*)(dk + 8) = *(const v4u*)(src + 8);
    }
    {  // V: b128 loads, transposed b16 scatter into [hd][key]
      const unsigned short* src = Vg + (size_t)(k0 + row) * HEAD_DIM + off;
      union { v4u u; unsigned short s[8]; } c0, c1;
      c0.u = *(const v4u*)(src);
      c1.u = *(const v4u*)(src + 8);
      unsigned short* dv = sV[buf];
#pragma unroll
      for (int i = 0; i < 8; ++i) dv[(off + i) * 72 + row] = c0.s[i];
#pragma unroll
      for (int i = 0; i < 8; ++i) dv[(off + 8 + i) * 72 + row] = c1.s[i];
    }
  };

  load_kv(0, 0);
  int cur = 0;
  for (int kt = 0; kt < nkv; ++kt) {
    int k0 = kt * 64;
    __syncthreads();                       // buf[cur] ready; buf[cur^1] free
    if (kt + 1 < nkv) load_kv(kt + 1, cur ^ 1);

    // S = Q @ K^T   (log2-scaled)
    v8f s[4];
#pragma unroll
    for (int j = 0; j < 4; ++j) {
      v16bf b0 = load_frag_b(sK[cur] + (j * 16) * 72 + 0, 72, lane);
      v16bf b1 = load_frag_b(sK[cur] + (j * 16) * 72 + 32, 72, lane);
      v8f a = zero8();
      a = wmma_bf16(qfr0, b0, a);
      a = wmma_bf16(qfr1, b1, a);
      s[j] = a;
    }

    // causal mask: only the last two kv tiles can straddle the diagonal
    if (k0 + 63 > q0) {                    // uniform scalar branch
      int kcol = k0 + (lane & 15);
#pragma unroll
      for (int j = 0; j < 4; ++j)
#pragma unroll
        for (int r = 0; r < 8; ++r)
          s[j][r] = (kcol + 16 * j > qrow_base + r) ? -3.0e38f : s[j][r];
    }

    // online softmax (f32, exp2 domain)
    float mnew[8];
#pragma unroll
    for (int r = 0; r < 8; ++r) mnew[r] = m_r[r];
#pragma unroll
    for (int j = 0; j < 4; ++j)
#pragma unroll
      for (int r = 0; r < 8; ++r) mnew[r] = fmaxf(mnew[r], s[j][r]);
#pragma unroll
    for (int msk = 1; msk <= 8; msk <<= 1)
#pragma unroll
      for (int r = 0; r < 8; ++r) mnew[r] = fmaxf(mnew[r], __shfl_xor(mnew[r], msk, 32));

    float rs[8];
#pragma unroll
    for (int r = 0; r < 8; ++r) rs[r] = 0.f;
#pragma unroll
    for (int j = 0; j < 4; ++j)
#pragma unroll
      for (int r = 0; r < 8; ++r) {
        float p = fast_exp2(s[j][r] - mnew[r]);
        s[j][r] = p;
        rs[r] += p;
      }
#pragma unroll
    for (int msk = 1; msk <= 8; msk <<= 1)
#pragma unroll
      for (int r = 0; r < 8; ++r) rs[r] += __shfl_xor(rs[r], msk, 32);

#pragma unroll
    for (int r = 0; r < 8; ++r) {
      float alpha = fast_exp2(m_r[r] - mnew[r]);
      l_r[r] = l_r[r] * alpha + rs[r];
      m_r[r] = mnew[r];
#pragma unroll
      for (int j = 0; j < 4; ++j) o_acc[j][r] *= alpha;
    }

    // P -> wave-private LDS region (C layout -> A layout)
    {
      unsigned short* pw = sP[wave];
      int rowoff = (lane >> 4) << 3;
      int col = lane & 15;
#pragma unroll
      for (int j = 0; j < 4; ++j)
#pragma unroll
        for (int r = 0; r < 8; ++r)
          pw[(rowoff + r) * 72 + j * 16 + col] = f2bf(s[j][r]);
    }
    // LDS is in-order per wave; wavefront fence orders store->load without
    // stalling on the (unrelated) next-tile prefetch like a block barrier would.
    __builtin_amdgcn_fence(__ATOMIC_ACQ_REL, "wavefront");

    // O += P @ V
    v16bf pa0 = load_frag_a(sP[wave] + 0, 72, lane);
    v16bf pa1 = load_frag_a(sP[wave] + 32, 72, lane);
#pragma unroll
    for (int j = 0; j < 4; ++j) {
      v16bf vb0 = load_frag_b(sV[cur] + (j * 16) * 72 + 0, 72, lane);
      v16bf vb1 = load_frag_b(sV[cur] + (j * 16) * 72 + 32, 72, lane);
      o_acc[j] = wmma_bf16(pa0, vb0, o_acc[j]);
      o_acc[j] = wmma_bf16(pa1, vb1, o_acc[j]);
    }
    cur ^= 1;
  }

  // finalize: O *= 1/l (v_rcp + mul, not 32 divides), write bf16 merged [B, L, D]
  float inv_l[8];
#pragma unroll
  for (int r = 0; r < 8; ++r) inv_l[r] = __builtin_amdgcn_rcpf(l_r[r]);
#pragma unroll
  for (int j = 0; j < 4; ++j)
#pragma unroll
    for (int r = 0; r < 8; ++r) {
      int q = qrow_base + r;
      int d = h * HEAD_DIM + j * 16 + (lane & 15);
      O[((size_t)b * SEQ + q) * N_EMBED + d] = f2bf(o_acc[j][r] * inv_l[r]);
    }
}

// ---------------------------------------------------------------------------
// Kernel 3: out = O_bf16 @ W_proj + b_proj  (f32 output), double-buffered
// ---------------------------------------------------------------------------
__global__ __launch_bounds__(256) void proj_gemm_kernel(
    const unsigned short* __restrict__ A, const float* __restrict__ W,
    const float* __restrict__ bias, float* __restrict__ out) {
  const int ldw = N_EMBED;
  int n0 = blockIdx.x * 128;
  int m0 = blockIdx.y * 128;
  int t = threadIdx.x, lane = t & 31, wave = t >> 5;
  int wm = wave & 3, wn = wave >> 2;
  __shared__ __align__(16) unsigned short sA[2][128 * 40];
  __shared__ __align__(16) unsigned short sB[2][128 * 40];
  v8f acc[2][4];
  for (int i = 0; i < 2; ++i)
    for (int j = 0; j < 4; ++j) acc[i][j] = zero8();

  auto load_tile = [&](int k0, int buf) {
    {  // A tile: already bf16, b128 copies
      int row = t >> 1, ko = (t & 1) * 16;
      const unsigned short* src = A + (size_t)(m0 + row) * N_EMBED + k0 + ko;
      unsigned short* dst = sA[buf] + row * 40 + ko;
      *(v4u*)(dst)     = *(const v4u*)(src);
      *(v4u*)(dst + 8) = *(const v4u*)(src + 8);
    }
    {  // B tile: W[k][n] f32 -> LDS [n][k] bf16, packed b32 stores
      int nn = t & 127, kb = (t >> 7) * 16;
      const float* wsrc = W + (size_t)(k0 + kb) * ldw + n0 + nn;
      unsigned int* dst = (unsigned int*)(sB[buf] + nn * 40 + kb);
#pragma unroll
      for (int i = 0; i < 8; ++i)
        dst[i] = f2bf_pk(wsrc[(size_t)(2 * i) * ldw], wsrc[(size_t)(2 * i + 1) * ldw]);
    }
  };

  load_tile(0, 0);
  int cur = 0;
  for (int k0 = 0; k0 < N_EMBED; k0 += 32) {
    __syncthreads();
    if (k0 + 32 < N_EMBED) load_tile(k0 + 32, cur ^ 1);
    v16bf a0 = load_frag_a(sA[cur] + (wm * 32) * 40, 40, lane);
    v16bf a1 = load_frag_a(sA[cur] + (wm * 32 + 16) * 40, 40, lane);
#pragma unroll
    for (int j = 0; j < 4; ++j) {
      v16bf bfr = load_frag_b(sB[cur] + (wn * 64 + j * 16) * 40, 40, lane);
      acc[0][j] = wmma_bf16(a0, bfr, acc[0][j]);
      acc[1][j] = wmma_bf16(a1, bfr, acc[1][j]);
    }
    cur ^= 1;
  }

  float bias_j[4];
#pragma unroll
  for (int j = 0; j < 4; ++j) bias_j[j] = bias[n0 + wn * 64 + j * 16 + (lane & 15)];
#pragma unroll
  for (int mi = 0; mi < 2; ++mi)
#pragma unroll
    for (int j = 0; j < 4; ++j)
#pragma unroll
      for (int r = 0; r < 8; ++r) {
        int m = m0 + wm * 32 + mi * 16 + r + ((lane >> 4) << 3);
        int n = n0 + wn * 64 + j * 16 + (lane & 15);
        out[(size_t)m * N_EMBED + n] = acc[mi][j][r] + bias_j[j];
      }
}

// ---------------------------------------------------------------------------
extern "C" void kernel_launch(void* const* d_in, const int* in_sizes, int n_in,
                              void* d_out, int out_size, void* d_ws, size_t ws_size,
                              hipStream_t stream) {
  const float* x      = (const float*)d_in[0];
  const float* W_attn = (const float*)d_in[1];
  const float* b_attn = (const float*)d_in[2];
  const float* W_proj = (const float*)d_in[3];
  const float* b_proj = (const float*)d_in[4];
  float* out = (float*)d_out;

  const size_t elems = (size_t)BATCH * SEQ * N_EMBED;  // 8,388,608
  unsigned short* Qw = (unsigned short*)d_ws;
  unsigned short* Kw = Qw + elems;
  unsigned short* Vw = Kw + elems;
  unsigned short* Ow = Vw + elems;

  qkv_gemm_kernel<<<dim3(3 * N_EMBED / 128, MROWS / 128), 256, 0, stream>>>(
      x, W_attn, b_attn, Qw, Kw, Vw);
  attn_kernel<<<dim3(SEQ / 128, BATCH * N_HEADS), 256, 0, stream>>>(Qw, Kw, Vw, Ow);
  proj_gemm_kernel<<<dim3(N_EMBED / 128, MROWS / 128), 256, 0, stream>>>(
      Ow, W_proj, b_proj, out);
}
